// DeformNet_21526376088359
// MI455X (gfx1250) — matrix operand order repaired and checked by reading, not compiled, not run on hardware
//
#include <hip/hip_runtime.h>

typedef __attribute__((ext_vector_type(16))) __bf16        v16bf;
typedef __attribute__((ext_vector_type(8)))  float         v8f;
typedef __attribute__((ext_vector_type(8)))  unsigned int  v8u;
typedef __attribute__((ext_vector_type(4)))  unsigned int  u32x4;
typedef __attribute__((ext_vector_type(8)))  int           i32x8;
typedef __attribute__((ext_vector_type(4)))  int           i32x4;

// ---------- sizes ----------
#define B4      4
#define C1      64
#define C2      128
#define H1      224
#define W1      224
#define H2      112
#define W2      112
#define PIX2    (H2*W2)            // 12544
#define NPIX    (B4*PIX2)          // 50176
#define K2      (C1*9)             // 576
#define K4      (C2*9)             // 1152
#define MT      4                  // M sub-tiles per block (64 pixels)

// ---------- workspace layout (byte offsets, all 256B aligned) ----------
#define OFF_H1    ((size_t)0)               // f32 4*64*224*224
#define OFF_H1P   ((size_t)51380224)        // bf16 4*64*112*112
#define OFF_Z     ((size_t)57802752)        // f32 4*128*112*112 (conv2+relu, pre-BN)
#define OFF_OFFS  ((size_t)83492864)        // f32 4*18*112*112
#define OFF_W2    ((size_t)87105536)        // bf16 128*576
#define OFF_W4    ((size_t)87252992)        // bf16 128*1152
#define OFF_S     ((size_t)87547904)        // f32 stats/affine block (768 floats)

__device__ __forceinline__ unsigned short f2bf(float f) {
    unsigned int u = __float_as_uint(f);
    u += 0x7fffu + ((u >> 16) & 1u);      // round-to-nearest-even
    return (unsigned short)(u >> 16);
}

// ---------- K0: weight conversion to bf16 + stats zeroing ----------
__global__ void k_prep(const float* __restrict__ c2w, const float* __restrict__ c4w,
                       unsigned short* __restrict__ w2bf, unsigned short* __restrict__ w4bf,
                       float* __restrict__ S) {
    int t = blockIdx.x * 256 + threadIdx.x;
    if (t < C2*K2)  w2bf[t] = f2bf(c2w[t]);
    if (t < C2*K4)  w4bf[t] = f2bf(c4w[t]);
    if (t < 768)    S[t] = 0.0f;
}

// ---------- K1: conv1 (3->64) + bias + relu, accumulate BN1 stats ----------
__global__ void k_conv1(const float* __restrict__ x, const float* __restrict__ w,
                        const float* __restrict__ bias, float* __restrict__ h1,
                        float* __restrict__ S) {
    __shared__ float ssum[256];
    __shared__ float ssq[256];
    int t   = blockIdx.x * 256 + threadIdx.x;          // < 4*64*224*224
    int c   = (t / (H1*W1)) % C1;
    int b   =  t / (C1*H1*W1);
    int rem =  t % (H1*W1);
    int oy  = rem / W1, ox = rem % W1;
    float acc = bias[c];
#pragma unroll
    for (int ci = 0; ci < 3; ++ci)
#pragma unroll
        for (int kh = 0; kh < 3; ++kh)
#pragma unroll
            for (int kw = 0; kw < 3; ++kw) {
                int iy = oy + kh - 1, ix = ox + kw - 1;
                if (iy >= 0 && iy < H1 && ix >= 0 && ix < W1)
                    acc += w[((c*3 + ci)*3 + kh)*3 + kw] *
                           x[((b*3 + ci)*H1 + iy)*W1 + ix];
            }
    acc = acc > 0.0f ? acc : 0.0f;
    h1[t] = acc;
    int tid = threadIdx.x;
    ssum[tid] = acc; ssq[tid] = acc * acc;
    __syncthreads();
    for (int s = 128; s > 0; s >>= 1) {
        if (tid < s) { ssum[tid] += ssum[tid + s]; ssq[tid] += ssq[tid + s]; }
        __syncthreads();
    }
    if (tid == 0) {                                     // block lies within one (b,c) plane
        atomicAdd(&S[c],      ssum[0]);
        atomicAdd(&S[64 + c], ssq[0]);
    }
}

// ---------- K2: finalize BN1 -> per-channel affine ----------
__global__ void k_fin1(float* __restrict__ S, const float* __restrict__ g,
                       const float* __restrict__ bb) {
    int c = threadIdx.x;                                // 64 threads
    float N    = (float)(B4*H1*W1);
    float mean = S[c] / N;
    float var  = S[64 + c] / N - mean * mean;
    float sc   = g[c] * rsqrtf(var + 1e-5f);
    S[384 + c] = sc;
    S[448 + c] = bb[c] - mean * sc;
}

// ---------- K3: pool(bn(h1)) == bn(pool(h1)) -> bf16 activations ----------
__global__ void k_poolbn(const float* __restrict__ h1, const float* __restrict__ S,
                         unsigned short* __restrict__ h1p) {
    int t   = blockIdx.x * 256 + threadIdx.x;           // < 4*64*112*112
    int c   = (t / PIX2) % C1;
    int b   =  t / (C1*PIX2);
    int rem =  t % PIX2;
    int oy  = rem / W2, ox = rem % W2;
    const float* p = h1 + ((size_t)(b*C1 + c)*H1 + oy*2)*W1 + ox*2;
    float v = 0.25f * (p[0] + p[1] + p[W1] + p[W1 + 1]);
    v = v * S[384 + c] + S[448 + c];
    h1p[t] = f2bf(v);
}

// ---------- K4: conv2 (64->128) implicit GEMM via WMMA bf16 ----------
// block = 256 (8 waves); 64-pixel tile (4 M sub-tiles), wave wv owns couts
// [16wv,16wv+16). The full bf16 B matrix (128x576 = 147KB) is DMA'd into LDS
// once per block by the Tensor Data Mover, overlapped with the im2col A fill;
// each B fragment is then reused by 4 register accumulators (4x arithmetic
// intensity on the L2 weight stream).
__global__ void k_conv2_wmma(const unsigned short* __restrict__ h1p,
                             const unsigned short* __restrict__ w2bf,
                             const float* __restrict__ bias,
                             float* __restrict__ z, float* __restrict__ S) {
    __shared__ unsigned short A[16*MT * 608];           // im2col tile, padded stride
    __shared__ unsigned short Bsh[C2 * K2];             // 147456 B weight tile
    __shared__ float s2sum[C2];
    __shared__ float s2sq[C2];
    const int tid = threadIdx.x;
    const int pixBase = blockIdx.x * (16*MT);
    if (tid < C2) { s2sum[tid] = 0.0f; s2sq[tid] = 0.0f; }

#if defined(__gfx1250__) && __has_builtin(__builtin_amdgcn_tensor_load_to_lds) && \
    __has_builtin(__builtin_amdgcn_s_wait_tensorcnt)
    const bool use_tdm = true;
    if (tid < 32) {                                     // wave 0 issues the TDM descriptor
        unsigned long long ga = (unsigned long long)(const void*)w2bf;
        unsigned int lds = (unsigned int)(size_t)&Bsh[0];   // low 32 bits = LDS offset
        u32x4 g0;
        g0[0] = 1u;                                     // count=1, user-mode, no gather
        g0[1] = lds;                                    // lds_addr
        g0[2] = (unsigned int)(ga & 0xffffffffu);       // global_addr[31:0]
        g0[3] = (unsigned int)((ga >> 32) & 0x01ffffffu) | (2u << 30); // addr[56:32] | type=2
        i32x8 g1;
        g1[0] = 0x00010000;                             // workgroup_mask=0, data_size=1 (2B)
        g1[1] = (int)((K2 & 0xffffu) << 16);            // tensor_dim0[15:0]
        g1[2] = (int)((K2 >> 16) | ((C2 & 0xffffu) << 16)); // dim0[31:16] | dim1[15:0]
        g1[3] = (int)((C2 >> 16) | ((unsigned)K2 << 16));   // dim1[31:16] | tile_dim0
        g1[4] = (int)(C2);                              // tile_dim1 | tile_dim2(=0)<<16
        g1[5] = (int)(K2);                              // tensor_dim0_stride[31:0]
        g1[6] = 0;                                      // stride0[47:32] | stride1[15:0]
        g1[7] = 0;                                      // stride1[47:16]
        i32x4 gz = {0, 0, 0, 0};                        // groups 2/3 unused (2D tensor)
#if __clang_major__ >= 23
        i32x8 gz8 = {0, 0, 0, 0, 0, 0, 0, 0};
        __builtin_amdgcn_tensor_load_to_lds(g0, g1, gz, gz, gz8, 0);
#else
        __builtin_amdgcn_tensor_load_to_lds(g0, g1, gz, gz, 0);
#endif
    }
#else
    const bool use_tdm = false;
#endif

    for (int i = tid; i < 16*MT * K2; i += 256) {       // im2col A tile (overlaps TDM)
        int m = i / K2, k = i - m * K2;
        int p = pixBase + m;
        int b = p / PIX2, rem = p - b * PIX2;
        int oy = rem / W2, ox = rem - oy * W2;
        int ci = k / 9, r9 = k - ci * 9;
        int kh = r9 / 3, kw = r9 - kh * 3;
        int iy = oy + kh - 1, ix = ox + kw - 1;
        unsigned short v = 0;
        if (iy >= 0 && iy < H2 && ix >= 0 && ix < W2)
            v = h1p[((b*C1 + ci)*H2 + iy)*W2 + ix];
        A[m * 608 + k] = v;
    }

    if (!use_tdm) {                                     // fallback: cooperative B load
        for (int i = tid; i < C2 * K2; i += 256)
            Bsh[i] = w2bf[i];
    }
#if defined(__gfx1250__) && __has_builtin(__builtin_amdgcn_tensor_load_to_lds) && \
    __has_builtin(__builtin_amdgcn_s_wait_tensorcnt)
    if (tid < 32) __builtin_amdgcn_s_wait_tensorcnt(0);
#endif
    __syncthreads();

    const int wv = tid >> 5, lane = tid & 31, l = lane & 15, hi = lane >> 4;
    const int coutBase = wv * 16;
    const unsigned int* brow = (const unsigned int*)&Bsh[(coutBase + l) * K2];
    v8f acc[MT];
#pragma unroll
    for (int mt = 0; mt < MT; ++mt) acc[mt] = (v8f){};
    for (int kb = 0; kb < K2; kb += 32) {
        v8u bu;
#pragma unroll
        for (int v = 0; v < 8; ++v)
            bu[v] = brow[(kb + ((v < 4) ? 2*v : 16 + 2*(v - 4)) + hi*8) >> 1]; // B 32x16 layout
        v16bf bv = __builtin_bit_cast(v16bf, bu);
#pragma unroll
        for (int mt = 0; mt < MT; ++mt) {
            const unsigned int* arow = (const unsigned int*)&A[(mt*16 + l) * 608];
            v8u au;
#pragma unroll
            for (int v = 0; v < 8; ++v)
                au[v] = arow[(kb + ((v < 4) ? 2*v : 16 + 2*(v - 4)) + hi*8) >> 1]; // A 16x32
            v16bf av = __builtin_bit_cast(v16bf, au);
            acc[mt] = __builtin_amdgcn_wmma_f32_16x16x32_bf16(false, av, false, bv,
                                                              (short)0, acc[mt], false, false);
        }
    }
#pragma unroll
    for (int mt = 0; mt < MT; ++mt)
#pragma unroll
        for (int r = 0; r < 8; ++r) {                    // D: M=r+8*hi, N=l
            int m = mt*16 + r + 8*hi, cout = coutBase + l;
            int p = pixBase + m;
            int b = p / PIX2, rem = p - b * PIX2;
            int oy = rem / W2, ox = rem - oy * W2;
            float v = acc[mt][r] + bias[cout];
            v = v > 0.0f ? v : 0.0f;
            z[((b*C2 + cout)*H2 + oy)*W2 + ox] = v;
            atomicAdd(&s2sum[cout], v);                  // ds_add_f32
            atomicAdd(&s2sq[cout],  v * v);
        }
    __syncthreads();
    if (tid < C2) {
        atomicAdd(&S[128 + tid], s2sum[tid]);
        atomicAdd(&S[256 + tid], s2sq[tid]);
    }
}

// ---------- K5: finalize BN2 -> per-channel affine ----------
__global__ void k_fin2(float* __restrict__ S, const float* __restrict__ g,
                       const float* __restrict__ bb) {
    int c = threadIdx.x;                                 // 128 threads
    float N    = (float)NPIX;
    float mean = S[128 + c] / N;
    float var  = S[256 + c] / N - mean * mean;
    float sc   = g[c] * rsqrtf(var + 1e-5f);
    S[512 + c] = sc;
    S[640 + c] = bb[c] - mean * sc;
}

// ---------- K6: offset conv (128->18), BN2 applied on the fly ----------
__global__ void k_offconv(const float* __restrict__ z, const float* __restrict__ S,
                          const float* __restrict__ w, const float* __restrict__ bias,
                          float* __restrict__ offs) {
    int t   = blockIdx.x * 256 + threadIdx.x;            // < 4*18*112*112
    int co  = (t / PIX2) % 18;
    int b   =  t / (18 * PIX2);
    int rem =  t % PIX2;
    int oy  = rem / W2, ox = rem % W2;
    float acc = bias[co];
    for (int ci = 0; ci < C2; ++ci) {
        float sc = S[512 + ci], sh = S[640 + ci];
        const float* zb = z + (size_t)(b*C2 + ci) * PIX2;
        const float* wr = w + (size_t)(co*C2 + ci) * 9;
#pragma unroll
        for (int kh = 0; kh < 3; ++kh)
#pragma unroll
            for (int kw = 0; kw < 3; ++kw) {
                int iy = oy + kh - 1, ix = ox + kw - 1;
                if (iy >= 0 && iy < H2 && ix >= 0 && ix < W2)
                    acc += wr[kh*3 + kw] * (zb[iy*W2 + ix]*sc + sh);
            }
    }
    offs[t] = acc;
}

// ---------- K7: deformable conv (128->128) via bilinear gather + WMMA ----------
// 64-pixel tile per block; B fragments from global (L2-resident weights) are
// each reused by 4 register accumulators.
__global__ void k_deform_wmma(const float* __restrict__ z, const float* __restrict__ S,
                              const float* __restrict__ offs,
                              const unsigned short* __restrict__ w4bf,
                              float* __restrict__ out) {
    __shared__ unsigned short A[16*MT * 1184];           // 64 x 1152 bf16 tile (padded)
    __shared__ float gw[16*MT * 9 * 4];
    __shared__ int   gidx[16*MT * 9 * 4];
    const int tid = threadIdx.x;
    const int pixBase = blockIdx.x * (16*MT);

    for (int t = tid; t < 16*MT * 9; t += 256) {         // per (pixel, sample) bilinear setup
        int m = t / 9, n = t - m * 9;
        int p = pixBase + m;
        int b = p / PIX2, rem = p - b * PIX2;
        int oy = rem / W2, ox = rem - oy * W2;
        float ofx = offs[((b*18 + n)*H2 + oy)*W2 + ox];
        float ofy = offs[((b*18 + 9 + n)*H2 + oy)*W2 + ox];
        float px = (float)(oy + 1 + n/3 - 1) + ofx;      // row coord in padded (114) space
        float py = (float)(ox + 1 + n%3 - 1) + ofy;      // col coord
        float fx = floorf(px), fy = floorf(py);
        float qlx = fminf(fmaxf(fx,        0.f), 113.f);
        float qly = fminf(fmaxf(fy,        0.f), 113.f);
        float qrx = fminf(fmaxf(fx + 1.f,  0.f), 113.f);
        float qry = fminf(fmaxf(fy + 1.f,  0.f), 113.f);
        bool  mx  = (px < 1.f) || (px > 112.f);
        bool  my  = (py < 1.f) || (py > 112.f);
        float pxc = fminf(fmaxf(mx ? fx : px, 0.f), 113.f);
        float pyc = fminf(fmaxf(my ? fy : py, 0.f), 113.f);
        float glt = (1.f + (qlx - pxc)) * (1.f + (qly - pyc));
        float grb = (1.f - (qrx - pxc)) * (1.f - (qry - pyc));
        float glb = (1.f + (qlx - pxc)) * (1.f - (qry - pyc));
        float grt = (1.f - (qrx - pxc)) * (1.f + (qly - pyc));
        int base = t * 4;
        float qxs[4] = {qlx, qrx, qlx, qrx};
        float qys[4] = {qly, qry, qry, qly};
        float gs [4] = {glt, grb, glb, grt};
#pragma unroll
        for (int j = 0; j < 4; ++j) {
            int iy2 = (int)qxs[j] - 1, ix2 = (int)qys[j] - 1;   // padded -> unpadded
            gidx[base + j] = (iy2 >= 0 && iy2 < H2 && ix2 >= 0 && ix2 < W2)
                           ? (iy2 * W2 + ix2) : -1;
            gw[base + j] = gs[j];
        }
    }
    __syncthreads();

    for (int i = tid; i < 16*MT * K4; i += 256) {        // build A tile (bilinear + BN2)
        int m = i / K4, k = i - m * K4;
        int ci = k / 9, n = k - ci * 9;
        int p = pixBase + m;
        int b = p / PIX2;
        const float* zb = z + (size_t)(b*C2 + ci) * PIX2;
        float sc = S[512 + ci], sh = S[640 + ci];
        int base = (m * 9 + n) * 4;
        float v = 0.0f;
#pragma unroll
        for (int j = 0; j < 4; ++j) {
            int idx = gidx[base + j];
            if (idx >= 0) v += gw[base + j] * (zb[idx] * sc + sh);
        }
        A[m * 1184 + k] = f2bf(v);
    }
    __syncthreads();

    const int wv = tid >> 5, lane = tid & 31, l = lane & 15, hi = lane >> 4;
    const int coutBase = wv * 16;
    const unsigned short* brow = w4bf + (size_t)(coutBase + l) * K4;
    __builtin_prefetch(brow, 0, 0);                      // global_prefetch_b8
    v8f acc[MT];
#pragma unroll
    for (int mt = 0; mt < MT; ++mt) acc[mt] = (v8f){};
    for (int kb = 0; kb < K4; kb += 32) {
        v8u bu;
#pragma unroll
        for (int v = 0; v < 8; ++v)
            bu[v] = *(const unsigned int*)(brow + kb + ((v < 4) ? 2*v : 16 + 2*(v - 4)) + hi*8);
        v16bf bv = __builtin_bit_cast(v16bf, bu);
#pragma unroll
        for (int mt = 0; mt < MT; ++mt) {
            const unsigned int* arow = (const unsigned int*)&A[(mt*16 + l) * 1184];
            v8u au;
#pragma unroll
            for (int v = 0; v < 8; ++v)
                au[v] = arow[(kb + ((v < 4) ? 2*v : 16 + 2*(v - 4)) + hi*8) >> 1];
            v16bf av = __builtin_bit_cast(v16bf, au);
            acc[mt] = __builtin_amdgcn_wmma_f32_16x16x32_bf16(false, av, false, bv,
                                                              (short)0, acc[mt], false, false);
        }
    }
#pragma unroll
    for (int mt = 0; mt < MT; ++mt)
#pragma unroll
        for (int r = 0; r < 8; ++r) {
            int m = mt*16 + r + 8*hi, cout = coutBase + l;
            int p = pixBase + m;
            int b = p / PIX2, rem = p - b * PIX2;
            int oy = rem / W2, ox = rem - oy * W2;
            out[((b*C2 + cout)*H2 + oy)*W2 + ox] = acc[mt][r];
        }
}

extern "C" void kernel_launch(void* const* d_in, const int* in_sizes, int n_in,
                              void* d_out, int out_size, void* d_ws, size_t ws_size,
                              hipStream_t stream) {
    const float* x    = (const float*)d_in[0];
    const float* c1w  = (const float*)d_in[1];
    const float* c1b  = (const float*)d_in[2];
    const float* bn1g = (const float*)d_in[3];
    const float* bn1b = (const float*)d_in[4];
    const float* c2w  = (const float*)d_in[5];
    const float* c2b  = (const float*)d_in[6];
    const float* bn2g = (const float*)d_in[7];
    const float* bn2b = (const float*)d_in[8];
    const float* offw = (const float*)d_in[9];
    const float* offb = (const float*)d_in[10];
    const float* c4w  = (const float*)d_in[11];

    char* ws = (char*)d_ws;
    float*          h1   = (float*)         (ws + OFF_H1);
    unsigned short* h1p  = (unsigned short*)(ws + OFF_H1P);
    float*          z    = (float*)         (ws + OFF_Z);
    float*          offs = (float*)         (ws + OFF_OFFS);
    unsigned short* w2bf = (unsigned short*)(ws + OFF_W2);
    unsigned short* w4bf = (unsigned short*)(ws + OFF_W4);
    float*          S    = (float*)         (ws + OFF_S);
    float* out = (float*)d_out;

    k_prep       <<<(C2*K4 + 255)/256, 256, 0, stream>>>(c2w, c4w, w2bf, w4bf, S);
    k_conv1      <<<(B4*C1*H1*W1)/256, 256, 0, stream>>>(x, c1w, c1b, h1, S);
    k_fin1       <<<1, 64, 0, stream>>>(S, bn1g, bn1b);
    k_poolbn     <<<(B4*C1*PIX2)/256, 256, 0, stream>>>(h1, S, h1p);
    k_conv2_wmma <<<NPIX/(16*MT), 256, 0, stream>>>(h1p, w2bf, c2b, z, S);
    k_fin2       <<<1, 128, 0, stream>>>(S, bn2g, bn2b);
    k_offconv    <<<(B4*18*PIX2)/256, 256, 0, stream>>>(z, S, offw, offb, offs);
    k_deform_wmma<<<NPIX/(16*MT), 256, 0, stream>>>(z, S, offs, w4bf, out);
}
